// CausalSelfAttention_16836271800822
// MI455X (gfx1250) — matrix operand order repaired and checked by reading, not compiled, not run on hardware
//
#include <hip/hip_runtime.h>
#include <hip/hip_bf16.h>

// Problem constants (match reference)
#define BB 2
#define TT 2048
#define DM 1024
#define NH 16
#define DH 64
#define F3 (3 * DM)          // 3072
#define MROWS (BB * TT)      // 4096

typedef __attribute__((ext_vector_type(16))) __bf16 v16bf;
typedef __attribute__((ext_vector_type(8)))  float  v8f;

union BF16Frag {
    v16bf    v;
    __bf16   b[16];
    unsigned u[8];
};

static __device__ inline v8f wmma_bf16(v16bf a, v16bf b, v8f c) {
    // D = A x B + C, f32 accumulate  -> v_wmma_f32_16x16x32_bf16
    return __builtin_amdgcn_wmma_f32_16x16x32_bf16(
        /*neg_a=*/false, a, /*neg_b=*/false, b,
        /*c_mod=*/(short)0, c, /*reuse_a=*/false, /*reuse_b=*/false);
}

// pack 4 fp32 -> 4 bf16 (compiler emits v_cvt_pk_bf16_f32 pairs)
static __device__ inline uint2 pack4_bf16(float4 f) {
    union { __bf16 b[4]; uint2 u; } t;
    t.b[0] = (__bf16)f.x; t.b[1] = (__bf16)f.y;
    t.b[2] = (__bf16)f.z; t.b[3] = (__bf16)f.w;
    return t.u;
}

static __device__ inline float red_max16(float v) {
    // reduce across the 16 lanes of a wave32 half (xor masks < 16 stay in-half)
    #pragma unroll
    for (int m = 1; m < 16; m <<= 1) v = fmaxf(v, __shfl_xor(v, m, 32));
    return v;
}
static __device__ inline float red_sum16(float v) {
    #pragma unroll
    for (int m = 1; m < 16; m <<= 1) v += __shfl_xor(v, m, 32);
    return v;
}

// -------------------------------------------------------------------------
// Kernel 1/4: C[M,N] = A[M,K] * B[N,K]^T   (fp32 in, bf16 WMMA, fp32 out)
// Workgroup: 256 threads (8 waves), block tile 128x64.
// Wave grid 4x2; each wave owns a 32x32 output tile:
//   2 A-frags x 2 B-frags -> 4 WMMAs per 32-wide K step (2x operand reuse).
// Staging: float4 global loads -> packed bf16 b64 LDS stores (deep MLP).
// M % 128 == 0, N % 64 == 0, K % 32 == 0 (true for all uses here).
// -------------------------------------------------------------------------
__global__ __launch_bounds__(256) void gemm_nt_bf16(
    const float* __restrict__ A, const float* __restrict__ B,
    float* __restrict__ C, int M, int N, int K)
{
    __shared__ __bf16 As[128][36];   // padded rows to dodge bank conflicts
    __shared__ __bf16 Bs[64][36];

    const int tid  = threadIdx.x;
    const int lane = tid & 31, wave = tid >> 5;
    const int wm   = wave >> 1, wn = wave & 1;       // 4x2 wave grid
    const int bm   = blockIdx.y * 128, bn = blockIdx.x * 64;
    const int l15  = lane & 15, hl = lane >> 4;

    // staging coordinates (contiguous 16B+ chunks per thread)
    const int ar = tid >> 1, ac0 = (tid & 1) * 16;   // A: 128 rows x 32
    const int br = tid >> 2, bc0 = (tid & 3) * 8;    // B:  64 rows x 32

    v8f z = {0.f,0.f,0.f,0.f,0.f,0.f,0.f,0.f};
    v8f acc[2][2];
    #pragma unroll
    for (int am = 0; am < 2; ++am)
        #pragma unroll
        for (int bn2 = 0; bn2 < 2; ++bn2) acc[am][bn2] = z;

    for (int k0 = 0; k0 < K; k0 += 32) {
        // ---- A tile: 4x b128 loads, 4x b64 LDS stores per thread ----
        {
            const float4* src = (const float4*)&A[(size_t)(bm + ar) * K + k0 + ac0];
            float4 f0 = src[0], f1 = src[1], f2 = src[2], f3 = src[3];
            *(uint2*)&As[ar][ac0 +  0] = pack4_bf16(f0);
            *(uint2*)&As[ar][ac0 +  4] = pack4_bf16(f1);
            *(uint2*)&As[ar][ac0 +  8] = pack4_bf16(f2);
            *(uint2*)&As[ar][ac0 + 12] = pack4_bf16(f3);
        }
        // ---- B tile: 2x b128 loads, 2x b64 LDS stores per thread ----
        {
            const float4* src = (const float4*)&B[(size_t)(bn + br) * K + k0 + bc0];
            float4 f0 = src[0], f1 = src[1];
            *(uint2*)&Bs[br][bc0 + 0] = pack4_bf16(f0);
            *(uint2*)&Bs[br][bc0 + 4] = pack4_bf16(f1);
        }
        // prefetch next K tile while this one is consumed (global_prefetch_b8)
        if (k0 + 32 < K) {
            __builtin_prefetch(&A[(size_t)(bm + (tid >> 1)) * K + k0 + 32], 0, 3);
            if (tid < 128)
                __builtin_prefetch(&B[(size_t)(bn + (tid >> 1)) * K + k0 + 32], 0, 3);
        }
        __syncthreads();

        // A fragments: 16x32, row = lane&15; k = (v/4)*16 + 8*half + 2*(v%4)+pos
        BF16Frag af[2];
        #pragma unroll
        for (int am = 0; am < 2; ++am)
            #pragma unroll
            for (int v = 0; v < 8; ++v) {
                int kk = ((v >> 2) << 4) + (hl << 3) + ((v & 3) << 1);
                af[am].u[v] = *(const unsigned*)&As[wm * 32 + am * 16 + l15][kk];
            }
        // B fragments: 32x16, col = lane&15; k = (v/4)*16 + 8*half + 2*(v%4)+pos
        BF16Frag bf[2];
        #pragma unroll
        for (int bn2 = 0; bn2 < 2; ++bn2)
            #pragma unroll
            for (int v = 0; v < 8; ++v) {
                int kk = ((v >> 2) << 4) + (hl << 3) + ((v & 3) << 1);
                bf[bn2].u[v] = *(const unsigned*)&Bs[wn * 32 + bn2 * 16 + l15][kk];
            }
        #pragma unroll
        for (int am = 0; am < 2; ++am)
            #pragma unroll
            for (int bn2 = 0; bn2 < 2; ++bn2)
                acc[am][bn2] = wmma_bf16(af[am].v, bf[bn2].v, acc[am][bn2]);
        __syncthreads();
    }

    // C layout: vgpr r -> row r + 8*half, col = lane&15
    #pragma unroll
    for (int am = 0; am < 2; ++am)
        #pragma unroll
        for (int r = 0; r < 8; ++r) {
            int m = bm + wm * 32 + am * 16 + r + 8 * hl;
            #pragma unroll
            for (int bn2 = 0; bn2 < 2; ++bn2)
                C[(size_t)m * N + bn + wn * 32 + bn2 * 16 + l15] = acc[am][bn2][r];
        }
}

// -------------------------------------------------------------------------
// Kernel 2/4: RoPE applied in-place to Q and K halves of the QKV buffer.
// One thread per complex pair. total = B*T * 2(parts) * NH * 32 pairs.
// -------------------------------------------------------------------------
__global__ void rope_kernel(float* __restrict__ qkv,
                            const float* __restrict__ freqs)
{
    int idx = blockIdx.x * blockDim.x + threadIdx.x;   // exact grid, no tail
    int i    =  idx        & 31;     // pair index within head
    int h    = (idx >> 5)  & 15;     // head
    int part = (idx >> 9)  & 1;      // 0 = q, 1 = k
    int bt   =  idx >> 10;           // 0 .. B*T-1
    int t    =  bt & (TT - 1);

    float ang = freqs[t * 32 + i];
    float c = __cosf(ang), s = __sinf(ang);
    size_t base = (size_t)bt * F3 + part * DM + h * DH + i * 2;
    float re = qkv[base], im = qkv[base + 1];
    qkv[base]     = re * c - im * s;
    qkv[base + 1] = re * s + im * c;
}

// -------------------------------------------------------------------------
// Kernel 3/4: causal flash attention.
// Grid: (B*NH, T/128). Block: 256 threads = 8 waves; wave owns 16 q-rows.
// K/V tiles of 32 keys staged in LDS (bf16); scores & PV via bf16 WMMA;
// online softmax in fp32 with 16-lane shuffles.
// -------------------------------------------------------------------------
__global__ __launch_bounds__(256) void flash_attn(
    const float* __restrict__ qkv, float* __restrict__ out)
{
    __shared__ __bf16 Ks[32][68];        // [key][d]
    __shared__ __bf16 VsT[64][36];       // [d][key]  (transposed for B-frag)
    __shared__ float  Ss[8][16][32];     // per-wave P staging (C -> A relayout)

    const int bh = blockIdx.x;
    const int b = bh >> 4, h = bh & 15;
    const int tid = threadIdx.x, lane = tid & 31, wave = tid >> 5;
    const int l15 = lane & 15, hl = lane >> 4;
    const int qblock = blockIdx.y * 128;
    const int qrow0  = qblock + wave * 16;
    const float scale = 0.125f;          // 1/sqrt(64)

    // staging coordinates: thread owns 8 contiguous d of one key row
    const int skk = tid >> 3, sc0 = (tid & 7) * 8;   // 32 keys x 64 d

    // Load the wave's Q fragments once (A layout, two K-steps over d=64)
    const float* qrow = qkv + (size_t)(b * TT + qrow0 + l15) * F3 + h * DH;
    BF16Frag qf[2];
    #pragma unroll
    for (int s = 0; s < 2; ++s)
        #pragma unroll
        for (int v = 0; v < 8; ++v) {
            int kk = s * 32 + ((v >> 2) << 4) + (hl << 3) + ((v & 3) << 1);
            float2 p = *(const float2*)&qrow[kk];
            qf[s].b[2 * v]     = (__bf16)p.x;
            qf[s].b[2 * v + 1] = (__bf16)p.y;
        }

    v8f z = {0.f,0.f,0.f,0.f,0.f,0.f,0.f,0.f};
    v8f O[4];
    #pragma unroll
    for (int dt = 0; dt < 4; ++dt) O[dt] = z;
    float mrow[8], lrow[8];
    #pragma unroll
    for (int r = 0; r < 8; ++r) { mrow[r] = -__builtin_inff(); lrow[r] = 0.f; }

    const int ktmax = (qblock + 127) >> 5;   // uniform across the workgroup

    for (int kt = 0; kt <= ktmax; ++kt) {
        const int kk0 = kt << 5;

        // ---- cooperative K/V tile load: batched b128 loads ----
        {
            const float* base =
                qkv + (size_t)(b * TT + kk0 + skk) * F3 + h * DH + sc0;
            const float4* ksrc = (const float4*)(base + DM);
            const float4* vsrc = (const float4*)(base + 2 * DM);
            float4 kf0 = ksrc[0], kf1 = ksrc[1];
            float4 vf0 = vsrc[0], vf1 = vsrc[1];
            *(uint2*)&Ks[skk][sc0 + 0] = pack4_bf16(kf0);
            *(uint2*)&Ks[skk][sc0 + 4] = pack4_bf16(kf1);
            // V transposed scatter (stores batched after both loads)
            VsT[sc0 + 0][skk] = (__bf16)vf0.x;
            VsT[sc0 + 1][skk] = (__bf16)vf0.y;
            VsT[sc0 + 2][skk] = (__bf16)vf0.z;
            VsT[sc0 + 3][skk] = (__bf16)vf0.w;
            VsT[sc0 + 4][skk] = (__bf16)vf1.x;
            VsT[sc0 + 5][skk] = (__bf16)vf1.y;
            VsT[sc0 + 6][skk] = (__bf16)vf1.z;
            VsT[sc0 + 7][skk] = (__bf16)vf1.w;
        }
        // prefetch next K/V tile (global_prefetch_b8)
        if (kt < ktmax && tid < 32) {
            const float* nkv =
                qkv + (size_t)(b * TT + kk0 + 32 + tid) * F3 + h * DH;
            __builtin_prefetch(nkv + DM, 0, 3);
            __builtin_prefetch(nkv + 2 * DM, 0, 3);
        }
        __syncthreads();

        // S = Q * K^T  (two 16x16 N-tiles, two K-steps over d)
        v8f S0 = z, S1 = z;
        #pragma unroll
        for (int s = 0; s < 2; ++s) {
            BF16Frag b0, b1;
            #pragma unroll
            for (int v = 0; v < 8; ++v) {
                int kk = s * 32 + ((v >> 2) << 4) + (hl << 3) + ((v & 3) << 1);
                b0.u[v] = *(const unsigned*)&Ks[l15][kk];
                b1.u[v] = *(const unsigned*)&Ks[16 + l15][kk];
            }
            S0 = wmma_bf16(qf[s].v, b0.v, S0);
            S1 = wmma_bf16(qf[s].v, b1.v, S1);
        }

        // mask + online softmax (row r+8*half lives in vgpr r of this lane)
        #pragma unroll
        for (int r = 0; r < 8; ++r) {
            int qi = qrow0 + r + 8 * hl;
            float s0 = (kk0 +      l15 <= qi) ? S0[r] * scale : -__builtin_inff();
            float s1 = (kk0 + 16 + l15 <= qi) ? S1[r] * scale : -__builtin_inff();
            float mx   = red_max16(fmaxf(s0, s1));
            float mnew = fmaxf(mrow[r], mx);
            float fac  = __expf(mrow[r] - mnew);   // 0 on first tile (m=-inf)
            float p0 = __expf(s0 - mnew);
            float p1 = __expf(s1 - mnew);
            S0[r] = p0; S1[r] = p1;
            lrow[r] = lrow[r] * fac + red_sum16(p0 + p1);
            mrow[r] = mnew;
            #pragma unroll
            for (int dt = 0; dt < 4; ++dt) O[dt][r] *= fac;
        }

        // Stage P through LDS to convert C-layout -> A-layout (bf16)
        #pragma unroll
        for (int r = 0; r < 8; ++r) {
            Ss[wave][r + 8 * hl][l15]      = S0[r];
            Ss[wave][r + 8 * hl][16 + l15] = S1[r];
        }
        asm volatile("s_wait_dscnt 0x0" ::: "memory");  // wave-private region

        BF16Frag pf;
        #pragma unroll
        for (int v = 0; v < 8; ++v) {
            int kk = ((v >> 2) << 4) + (hl << 3) + ((v & 3) << 1);
            pf.b[2 * v]     = (__bf16)Ss[wave][l15][kk];
            pf.b[2 * v + 1] = (__bf16)Ss[wave][l15][kk + 1];
        }

        // O += P * V   (4 d-tiles of 16, K=32)
        #pragma unroll
        for (int dt = 0; dt < 4; ++dt) {
            BF16Frag vf;
            #pragma unroll
            for (int v = 0; v < 8; ++v) {
                int kk = ((v >> 2) << 4) + (hl << 3) + ((v & 3) << 1);
                vf.u[v] = *(const unsigned*)&VsT[dt * 16 + l15][kk];
            }
            O[dt] = wmma_bf16(pf.v, vf.v, O[dt]);
        }
        __syncthreads();
    }

    // epilogue: O /= l, write [b,t,h,d] -> flat [bt, 1024]
    #pragma unroll
    for (int r = 0; r < 8; ++r) {
        float inv = 1.0f / lrow[r];
        int qi = qrow0 + r + 8 * hl;
        size_t base = (size_t)(b * TT + qi) * DM + h * DH;
        #pragma unroll
        for (int dt = 0; dt < 4; ++dt)
            out[base + dt * 16 + l15] = O[dt][r] * inv;
    }
}

// -------------------------------------------------------------------------
// Launch: qkv GEMM -> rope -> flash attention -> output GEMM
// d_in: x (fp32), rope_freqs (fp32), w_qkv (fp32), w_out (fp32)
// d_ws: [0,48MB) qkv buffer, [48MB,64MB) attention output
// -------------------------------------------------------------------------
extern "C" void kernel_launch(void* const* d_in, const int* in_sizes, int n_in,
                              void* d_out, int out_size, void* d_ws, size_t ws_size,
                              hipStream_t stream) {
    const float* x      = (const float*)d_in[0];
    const float* freqs  = (const float*)d_in[1];
    const float* w_qkv  = (const float*)d_in[2];
    const float* w_out  = (const float*)d_in[3];
    float* out = (float*)d_out;

    float* qkv_ws  = (float*)d_ws;                       // 4096*3072 f32
    float* attn_ws = qkv_ws + (size_t)MROWS * F3;        // 4096*1024 f32

    // 1) QKV = X * Wqkv^T : M=4096, N=3072, K=1024
    gemm_nt_bf16<<<dim3(F3 / 64, MROWS / 128), 256, 0, stream>>>(
        x, w_qkv, qkv_ws, MROWS, F3, DM);

    // 2) RoPE on q,k (B*T * 2 * 16 heads * 32 pairs threads)
    rope_kernel<<<(MROWS * 2 * NH * 32) / 256, 256, 0, stream>>>(qkv_ws, freqs);

    // 3) causal flash attention
    flash_attn<<<dim3(BB * NH, TT / 128), 256, 0, stream>>>(qkv_ws, attn_ws);

    // 4) OUT = Attn * Wout^T : M=4096, N=1024, K=1024
    gemm_nt_bf16<<<dim3(DM / 64, MROWS / 128), 256, 0, stream>>>(
        attn_ws, w_out, out, MROWS, DM, DM);
}
